// NetModel_75307956568280
// MI455X (gfx1250) — matrix operand order repaired and checked by reading, not compiled, not run on hardware
//
#include <hip/hip_runtime.h>

typedef __attribute__((ext_vector_type(16))) _Float16 v16h;
typedef __attribute__((ext_vector_type(8)))  _Float16 v8h;
typedef __attribute__((ext_vector_type(8)))  float    v8f;

// ---------------- problem constants ----------------
#define CN_N    88863
#define CN_K    10
#define CN_B    128
#define CN_D    500
#define CN_ED   172
#define CN_RAW  1672
#define CN_HID  836
#define CN_MSG  100
#define CN_NM   20000
#define CN_QD   1000
#define CN_KD   1172
#define CN_MQ1  384      // 3*B queries, layer 1
#define CN_MQ0  3840     // MQ1*K queries, layer 0
#define CN_R0   38400    // MQ0*K neighbor rows, layer 0

#define AS_STR  40       // LDS row stride in halfs (32 data + 8 pad), 80B (16B aligned)

// =====================================================================
// WMMA GEMM:  C[M,N](ldc) = act(A[M,K](lda) @ W[K,N] + bias)
// f32 in/out, f16 multiply, f32 accumulate.
// Block = 256 threads = 8 waves -> 256x64 tile; wave -> 32x64 (8 WMMAs/step,
// 2 A-fragments share each B-fragment).
// A (256x32) and B^T (64x32) staged in LDS per K-step, converted to f16 once.
// Tail path: batched clamped in-bounds loads * {1,0} selector (no predication,
// one wait for 8 loads).
// =====================================================================
template<int RELU>
__global__ __launch_bounds__(256)
void gemm_wmma_f16(const float* __restrict__ A, const float* __restrict__ W,
                   const float* __restrict__ bias, float* __restrict__ C,
                   int M, int N, int K, int lda, int ldc)
{
  __shared__ __align__(16) _Float16 As[256 * AS_STR];   // [row][k]
  __shared__ __align__(16) _Float16 Bt[64 * AS_STR];    // [n][k]  (transposed)

  const int tid  = threadIdx.x;
  const int lane = tid & 31;
  const int wave = tid >> 5;
  const int m0b  = blockIdx.y * 256;
  const int n0   = blockIdx.x * 64;
  const int half = lane >> 4;   // 0/1
  const int l15  = lane & 15;

  // staging assignments: A -> one full 32-col row per thread; B -> col bn, 8 k's
  const int  arow_g = m0b + tid;
  const int  arow_c = (arow_g < M) ? arow_g : (M - 1);
  const float am    = (arow_g < M) ? 1.f : 0.f;
  const int  bn     = tid & 63;
  const int  bk     = (tid >> 6) * 8;
  const int  bcol_g = n0 + bn;
  const int  bcol_c = (bcol_g < N) ? bcol_g : (N - 1);
  const float bm    = (bcol_g < N) ? 1.f : 0.f;

  const bool edgeMN = (m0b + 256 > M) || (n0 + 64 > N);

  v8f z = {0.f,0.f,0.f,0.f,0.f,0.f,0.f,0.f};
  v8f acc[2][4] = {{z, z, z, z}, {z, z, z, z}};

  for (int kk = 0; kk < K; kk += 32) {
    const bool fast = !edgeMN && (kk + 32 <= K);   // uniform across block
    if (fast) {
      // ---- A: 32 contiguous floats -> 32 halfs (four b128 LDS stores)
      const float4* ap4 = (const float4*)(A + (size_t)arow_g * lda + kk);
      float4 f[8];
#pragma unroll
      for (int i = 0; i < 8; ++i) f[i] = ap4[i];
#pragma unroll
      for (int p = 0; p < 4; ++p) {
        v8h h;
        h[0]=(_Float16)f[2*p].x;   h[1]=(_Float16)f[2*p].y;
        h[2]=(_Float16)f[2*p].z;   h[3]=(_Float16)f[2*p].w;
        h[4]=(_Float16)f[2*p+1].x; h[5]=(_Float16)f[2*p+1].y;
        h[6]=(_Float16)f[2*p+1].z; h[7]=(_Float16)f[2*p+1].w;
        *(v8h*)&As[tid * AS_STR + p * 8] = h;
      }
      // ---- B: 8 k-rows of this thread's column (coalesced across 64 lanes/row)
      const float* wp = W + (size_t)(kk + bk) * N + bcol_g;
      v8h hb;
#pragma unroll
      for (int i = 0; i < 8; ++i) hb[i] = (_Float16)wp[(size_t)i * N];
      *(v8h*)&Bt[bn * AS_STR + bk] = hb;
      if (kk + 64 < K) {
        __builtin_prefetch(A + (size_t)arow_g * lda + kk + 32, 0, 1);
        __builtin_prefetch(W + (size_t)(kk + 32 + bk) * N + bcol_g, 0, 1);
      }
    } else {
      // ---- branchless tail: batch 8 clamped loads, then convert (single wait)
      const float* arp = A + (size_t)arow_c * lda;
#pragma unroll
      for (int p = 0; p < 4; ++p) {
        float xv[8];
#pragma unroll
        for (int i = 0; i < 8; ++i) {
          const int k  = kk + p * 8 + i;
          const int kc = (k < K) ? k : (K - 1);
          xv[i] = arp[kc];
        }
        v8h h;
#pragma unroll
        for (int i = 0; i < 8; ++i) {
          const int k  = kk + p * 8 + i;
          const float sel = (k < K) ? am : 0.f;
          h[i] = (_Float16)(xv[i] * sel);
        }
        *(v8h*)&As[tid * AS_STR + p * 8] = h;
      }
      float bx[8];
#pragma unroll
      for (int i = 0; i < 8; ++i) {
        const int k  = kk + bk + i;
        const int kc = (k < K) ? k : (K - 1);
        bx[i] = W[(size_t)kc * N + bcol_c];
      }
      v8h hb;
#pragma unroll
      for (int i = 0; i < 8; ++i) {
        const int k  = kk + bk + i;
        const float sel = (k < K) ? bm : 0.f;
        hb[i] = (_Float16)(bx[i] * sel);
      }
      *(v8h*)&Bt[bn * AS_STR + bk] = hb;
    }
    __syncthreads();

    // ---- A fragments (16-bit A 16x32 layout): j0..7 -> k=half*8+j, j8..15 -> k=16+half*8+j
    v16h a[2];
#pragma unroll
    for (int u = 0; u < 2; ++u) {
      const int mloc = wave * 32 + u * 16 + l15;
      const v8h a0 = *(const v8h*)&As[mloc * AS_STR + half * 8];
      const v8h a1 = *(const v8h*)&As[mloc * AS_STR + 16 + half * 8];
      a[u] = __builtin_shufflevector(a0, a1, 0,1,2,3,4,5,6,7,8,9,10,11,12,13,14,15);
    }
    // ---- 4 B fragments (B 32x16 layout: j0..7 -> k=half*8+j, j8..15 -> k=16+half*8+j), each feeds 2 WMMAs
#pragma unroll
    for (int t = 0; t < 4; ++t) {
      const int nloc = t * 16 + l15;
      const v8h b0 = *(const v8h*)&Bt[nloc * AS_STR + half * 8];
      const v8h b1 = *(const v8h*)&Bt[nloc * AS_STR + 16 + half * 8];
      const v16h b = __builtin_shufflevector(b0, b1, 0,1,2,3,4,5,6,7,8,9,10,11,12,13,14,15);
      acc[0][t] = __builtin_amdgcn_wmma_f32_16x16x32_f16(
          false, a[0], false, b, (short)0, acc[0][t], false, false);
      acc[1][t] = __builtin_amdgcn_wmma_f32_16x16x32_f16(
          false, a[1], false, b, (short)0, acc[1][t], false, false);
    }
    __syncthreads();
  }

  // ---- epilogue: D layout: VGPR r -> M = m0 + half*8 + r, N = n0 + t*16 + l15
#pragma unroll
  for (int u = 0; u < 2; ++u) {
    const int m0 = m0b + wave * 32 + u * 16;
#pragma unroll
    for (int t = 0; t < 4; ++t) {
      const int n = n0 + t * 16 + l15;
      if (n < N) {
        const float bv = bias ? bias[n] : 0.f;
#pragma unroll
        for (int r = 0; r < 8; ++r) {
          const int m = m0 + (half << 3) + r;
          if (m < M) {
            float x = acc[u][t][r] + bv;
            if (RELU) x = fmaxf(x, 0.f);
            C[(size_t)m * ldc + n] = x;
          }
        }
      }
    }
  }
}

// =====================================================================
// Row gather (memory[msg_nodes] -> h_g), 16B chunks.
// Uses CDNA5 async global->LDS copy + s_wait_asynccnt when available.
// =====================================================================
__global__ __launch_bounds__(256)
void k_gather_rows16(float* __restrict__ dst, const float* __restrict__ src,
                     const int* __restrict__ idx, long nchunks, int chunksPerRow) {
  long i = (long)blockIdx.x * 256 + threadIdx.x;
  const bool ok = (i < nchunks);
  const long ic = ok ? i : 0;
  const int r = (int)(ic / chunksPerRow);
  const int c = (int)(ic % chunksPerRow);
  const float4* gsrc = (const float4*)(src + (size_t)idx[r] * CN_D) + c;
  float4*       gdst = (float4*)(dst + (size_t)r * CN_D) + c;
#if __has_builtin(__builtin_amdgcn_global_load_async_to_lds_b128)
  __shared__ __align__(16) char buf[256 * 16];
  typedef int int4v __attribute__((ext_vector_type(4)));
  typedef __attribute__((address_space(1))) int4v* gptr_t;
  typedef __attribute__((address_space(3))) int4v* lptr_t;
  gptr_t gp = (gptr_t)(const void*)gsrc;
  lptr_t lp = (lptr_t)(void*)&buf[threadIdx.x * 16];
  __builtin_amdgcn_global_load_async_to_lds_b128(gp, lp, 0, 0);
#if __has_builtin(__builtin_amdgcn_s_wait_asynccnt)
  __builtin_amdgcn_s_wait_asynccnt(0);
#else
  asm volatile("s_wait_asynccnt 0x0" ::: "memory");
#endif
  const float4 v = *(const float4*)&buf[threadIdx.x * 16];
  if (ok) *gdst = v;
#else
  if (ok) *gdst = *gsrc;
#endif
}

// =====================================================================
// Elementwise / gather kernels
// =====================================================================
__global__ void k_init_map(int* map, int n) {
  int i = blockIdx.x * blockDim.x + threadIdx.x;
  if (i < n) map[i] = -1;
}
__global__ void k_scatter_map(int* map, const int* __restrict__ msg_nodes, int nm) {
  int i = blockIdx.x * blockDim.x + threadIdx.x;
  if (i < nm) map[msg_nodes[i]] = i;
}
__global__ void k_gru(const float* __restrict__ gi, const float* __restrict__ gh,
                      const float* __restrict__ h, float* __restrict__ memup) {
  long i = (long)blockIdx.x * blockDim.x + threadIdx.x;
  long tot = (long)CN_NM * CN_D;
  if (i >= tot) return;
  int r = (int)(i / CN_D), d = (int)(i % CN_D);
  const float* gir = gi + (size_t)r * 1500;
  const float* ghr = gh + (size_t)r * 1500;
  float rr = 1.f / (1.f + expf(-(gir[d] + ghr[d])));
  float zz = 1.f / (1.f + expf(-(gir[500 + d] + ghr[500 + d])));
  float nn = tanhf(gir[1000 + d] + rr * ghr[1000 + d]);
  memup[i] = (1.f - zz) * nn + zz * h[i];
}
__device__ __forceinline__ float feat_at(int node, int d,
    const float* __restrict__ node_raw, const float* __restrict__ memory,
    const float* __restrict__ memup, const int* __restrict__ map) {
  int s = map[node];
  float mm = (s >= 0) ? memup[(size_t)s * CN_D + d] : memory[(size_t)node * CN_D + d];
  return node_raw[(size_t)node * CN_D + d] + mm;
}
__global__ void k_build_nodes(int* __restrict__ nodes, float* __restrict__ tsq,
                              const int* __restrict__ src, const int* __restrict__ dst,
                              const int* __restrict__ neg, const float* __restrict__ et) {
  int i = blockIdx.x * blockDim.x + threadIdx.x;
  if (i >= 3 * CN_B) return;
  nodes[i] = (i < CN_B) ? src[i] : (i < 2 * CN_B) ? dst[i - CN_B] : neg[i - 2 * CN_B];
  tsq[i] = et[i % CN_B];
}
__global__ void k_gather_nbrs(const int* __restrict__ q, int rows,
                              const int* __restrict__ nbr_nodes, const int* __restrict__ nbr_eidx,
                              const float* __restrict__ nbr_times,
                              int* __restrict__ nf, int* __restrict__ ef, float* __restrict__ tf) {
  int i = blockIdx.x * blockDim.x + threadIdx.x;
  if (i >= rows) return;
  size_t idx = (size_t)q[i / CN_K] * CN_K + (i % CN_K);
  nf[i] = nbr_nodes[idx];
  ef[i] = nbr_eidx[idx];
  tf[i] = nbr_times[idx];
}
__global__ void k_build_qin(float* __restrict__ qin, const int* __restrict__ qnodes, int Mq,
    const float* __restrict__ node_raw, const float* __restrict__ memory,
    const float* __restrict__ memup, const int* __restrict__ map,
    const float* __restrict__ time_b) {
  long i = (long)blockIdx.x * blockDim.x + threadIdx.x;
  long tot = (long)Mq * CN_QD;
  if (i >= tot) return;
  int r = (int)(i / CN_QD), c = (int)(i % CN_QD);
  qin[i] = (c < CN_D) ? feat_at(qnodes[r], c, node_raw, memory, memup, map)
                      : cosf(time_b[c - CN_D]);
}
__global__ void k_build_kin(float* __restrict__ kin, int rows,
    const int* __restrict__ nf, const int* __restrict__ ef,
    const float* __restrict__ tn, const float* __restrict__ tq,
    const float* __restrict__ nbrMat,
    const float* __restrict__ edge_raw, const float* __restrict__ time_w,
    const float* __restrict__ time_b,
    const float* __restrict__ node_raw, const float* __restrict__ memory,
    const float* __restrict__ memup, const int* __restrict__ map) {
  long i = (long)blockIdx.x * blockDim.x + threadIdx.x;
  long tot = (long)rows * CN_KD;
  if (i >= tot) return;
  int r = (int)(i / CN_KD), c = (int)(i % CN_KD);
  float v;
  if (c < CN_D) {
    v = nbrMat ? nbrMat[(size_t)r * CN_D + c]
               : feat_at(nf[r], c, node_raw, memory, memup, map);
  } else if (c < CN_D + CN_ED) {
    v = edge_raw[(size_t)ef[r] * CN_ED + (c - CN_D)];
  } else {
    int d = c - (CN_D + CN_ED);
    float dt = tq[r / CN_K] - tn[r];
    v = cosf(dt * time_w[d] + time_b[d]);
  }
  kin[i] = v;
}
// one wave per (query, head): 10 dots of length 500, softmax, weighted sum
__global__ __launch_bounds__(256)
void k_attn(const float* __restrict__ q, const float* __restrict__ kmat,
            const float* __restrict__ vmat, const int* __restrict__ nbr,
            float* __restrict__ o, int Mq) {
  const int lane = threadIdx.x & 31;
  const int gw = blockIdx.x * 8 + (threadIdx.x >> 5);
  const int m = gw >> 1;
  const int h = gw & 1;
  if (m >= Mq) return;
  const float* qrow = q + (size_t)m * CN_QD + h * CN_D;
  float s[CN_K];
#pragma unroll
  for (int kk = 0; kk < CN_K; ++kk) {
    const float* krow = kmat + (size_t)(m * CN_K + kk) * CN_QD + h * CN_D;
    float p = 0.f;
    for (int d = lane; d < CN_D; d += 32) p += qrow[d] * krow[d];
#pragma unroll
    for (int off = 16; off > 0; off >>= 1) p += __shfl_xor(p, off, 32);
    p *= 0.04472135954999579f;  // 1/sqrt(500)
    s[kk] = (nbr[m * CN_K + kk] == 0) ? -1.0e9f : p;
  }
  float mx = s[0];
#pragma unroll
  for (int kk = 1; kk < CN_K; ++kk) mx = fmaxf(mx, s[kk]);
  float sum = 0.f;
#pragma unroll
  for (int kk = 0; kk < CN_K; ++kk) { s[kk] = expf(s[kk] - mx); sum += s[kk]; }
  const float inv = 1.f / sum;
  for (int d = lane; d < CN_D; d += 32) {
    float accv = 0.f;
#pragma unroll
    for (int kk = 0; kk < CN_K; ++kk)
      accv += s[kk] * vmat[(size_t)(m * CN_K + kk) * CN_QD + h * CN_D + d];
    o[(size_t)m * CN_QD + h * CN_D + d] = accv * inv;
  }
}
__global__ void k_copy_cols(float* __restrict__ dst, int ldd, int colOff,
                            const float* __restrict__ src, int lds_, int cols, int rows) {
  long i = (long)blockIdx.x * blockDim.x + threadIdx.x;
  long tot = (long)rows * cols;
  if (i >= tot) return;
  int r = (int)(i / cols), c = (int)(i % cols);
  dst[(size_t)r * ldd + colOff + c] = src[(size_t)r * lds_ + c];
}
__global__ void k_build_aff_in(float* __restrict__ X, const float* __restrict__ emb) {
  long i = (long)blockIdx.x * blockDim.x + threadIdx.x;
  long tot = (long)2 * CN_B * CN_QD;
  if (i >= tot) return;
  int r = (int)(i / CN_QD), c = (int)(i % CN_QD);
  X[i] = (c < CN_D) ? emb[(size_t)(r & (CN_B - 1)) * CN_D + c]
                    : emb[(size_t)(CN_B + r) * CN_D + (c - CN_D)];
}
// final GEMV + sigmoid: one wave per row
__global__ __launch_bounds__(256)
void k_aff_out(const float* __restrict__ haff, const float* __restrict__ w2,
               const float* __restrict__ b2, float* __restrict__ out, int rows) {
  const int lane = threadIdx.x & 31;
  const int row = blockIdx.x * 8 + (threadIdx.x >> 5);
  if (row >= rows) return;
  float p = 0.f;
  for (int d = lane; d < CN_D; d += 32) p += haff[(size_t)row * CN_D + d] * w2[d];
#pragma unroll
  for (int off = 16; off > 0; off >>= 1) p += __shfl_xor(p, off, 32);
  if (lane == 0) out[row] = 1.f / (1.f + expf(-(p + b2[0])));
}

// =====================================================================
static void launch_gemm(hipStream_t s, const float* A, const float* W, const float* bias,
                        float* C, int M, int N, int K, int lda, int ldc, bool relu) {
  dim3 grid((unsigned)((N + 63) / 64), (unsigned)((M + 255) / 256));
  if (relu) gemm_wmma_f16<1><<<grid, 256, 0, s>>>(A, W, bias, C, M, N, K, lda, ldc);
  else      gemm_wmma_f16<0><<<grid, 256, 0, s>>>(A, W, bias, C, M, N, K, lda, ldc);
}
static inline unsigned cdiv(long a, long b) { return (unsigned)((a + b - 1) / b); }

extern "C" void kernel_launch(void* const* d_in, const int* in_sizes, int n_in,
                              void* d_out, int out_size, void* d_ws, size_t ws_size,
                              hipStream_t stream) {
  (void)in_sizes; (void)n_in; (void)out_size; (void)ws_size;
  const int*   src_nodes = (const int*)  d_in[0];
  const int*   dst_nodes = (const int*)  d_in[1];
  const int*   neg_nodes = (const int*)  d_in[2];
  const float* edge_times= (const float*)d_in[3];
  const int*   nbr_nodes = (const int*)  d_in[4];
  const int*   nbr_eidx  = (const int*)  d_in[5];
  const float* nbr_times = (const float*)d_in[6];
  const int*   msg_nodes = (const int*)  d_in[7];
  const float* messages  = (const float*)d_in[8];
  const float* memory    = (const float*)d_in[9];
  const float* node_raw  = (const float*)d_in[10];
  const float* edge_raw  = (const float*)d_in[11];
  const float* time_w    = (const float*)d_in[12];
  const float* time_b    = (const float*)d_in[13];
  const float* msg_W1    = (const float*)d_in[14];
  const float* msg_b1    = (const float*)d_in[15];
  const float* msg_W2    = (const float*)d_in[16];
  const float* msg_b2    = (const float*)d_in[17];
  const float* gru_Wih   = (const float*)d_in[18];
  const float* gru_Whh   = (const float*)d_in[19];
  const float* gru_bih   = (const float*)d_in[20];
  const float* gru_bhh   = (const float*)d_in[21];
  const float* att_Wq    = (const float*)d_in[22];
  const float* att_bq    = (const float*)d_in[23];
  const float* att_Wk    = (const float*)d_in[24];
  const float* att_bk    = (const float*)d_in[25];
  const float* att_Wv    = (const float*)d_in[26];
  const float* att_bv    = (const float*)d_in[27];
  const float* att_Wo    = (const float*)d_in[28];
  const float* att_bo    = (const float*)d_in[29];
  const float* mrg_W1    = (const float*)d_in[30];
  const float* mrg_b1    = (const float*)d_in[31];
  const float* mrg_W2    = (const float*)d_in[32];
  const float* mrg_b2    = (const float*)d_in[33];
  const float* aff_W1    = (const float*)d_in[34];
  const float* aff_b1    = (const float*)d_in[35];
  const float* aff_W2    = (const float*)d_in[36];
  const float* aff_b2    = (const float*)d_in[37];
  float* out = (float*)d_out;

  // ---------------- workspace layout ----------------
  char* wsb = (char*)d_ws;
  size_t cur = 0;
  auto alloc = [&](size_t nbytes) -> void* {
    void* p = wsb + cur;
    cur += (nbytes + 255) & ~(size_t)255;
    return p;
  };
  // persistent
  int*   map     = (int*)  alloc((size_t)CN_N * 4);
  float* mem_upd = (float*)alloc((size_t)CN_NM * CN_D * 4);
  int*   nodes   = (int*)  alloc(CN_MQ1 * 4);
  float* tsq     = (float*)alloc(CN_MQ1 * 4);
  int*   n1f     = (int*)  alloc(CN_MQ0 * 4);
  int*   e1f     = (int*)  alloc(CN_MQ0 * 4);
  float* t1f     = (float*)alloc(CN_MQ0 * 4);
  int*   n2f     = (int*)  alloc(CN_R0 * 4);
  int*   e2f     = (int*)  alloc(CN_R0 * 4);
  float* t2f     = (float*)alloc(CN_R0 * 4);
  float* qin0    = (float*)alloc((size_t)CN_MQ0 * CN_QD * 4);
  float* qin1    = (float*)alloc((size_t)CN_MQ1 * CN_QD * 4);
  float* emb1    = (float*)alloc((size_t)CN_MQ0 * CN_D * 4);
  float* emb     = (float*)alloc((size_t)CN_MQ1 * CN_D * 4);
  float* Xaff    = (float*)alloc((size_t)2 * CN_B * CN_QD * 4);
  float* haff    = (float*)alloc((size_t)2 * CN_B * CN_D * 4);
  // scratch arena (phase-overlapped)
  float* S = (float*)(wsb + cur);
  // phase A (message MLP + GRU)
  float* h_g  = S;
  float* hid  = h_g  + (size_t)CN_NM * CN_D;
  float* mmsg = hid  + (size_t)CN_NM * CN_HID;
  float* gi   = mmsg + (size_t)CN_NM * CN_MSG;
  float* gh   = gi   + (size_t)CN_NM * 1500;
  // phase B0 (layer-0 attention) - overlaps phase A
  float* kin0  = S;
  float* kmat0 = kin0  + (size_t)CN_R0 * CN_KD;
  float* vmat0 = kmat0 + (size_t)CN_R0 * CN_QD;
  float* q0    = vmat0 + (size_t)CN_R0 * CN_QD;
  float* o0    = S;                                 // reuse kin0 space (dead by then)
  float* hcat0 = o0    + (size_t)CN_MQ0 * CN_QD;
  float* mrgh0 = hcat0 + (size_t)CN_MQ0 * 1500;
  // phase B1 (layer-1 attention) - overlaps again
  float* kin1  = S;
  float* kmat1 = kin1  + (size_t)CN_MQ0 * CN_KD;
  float* vmat1 = kmat1 + (size_t)CN_MQ0 * CN_QD;
  float* q1    = vmat1 + (size_t)CN_MQ0 * CN_QD;
  float* o1    = q1    + (size_t)CN_MQ1 * CN_QD;
  float* hcat1 = o1    + (size_t)CN_MQ1 * CN_QD;
  float* mrgh1 = hcat1 + (size_t)CN_MQ1 * 1500;

  // ============ Phase A: messages -> GRU memory update ============
  launch_gemm(stream, messages, msg_W1, msg_b1, hid, CN_NM, CN_HID, CN_RAW, CN_RAW, CN_HID, true);
  launch_gemm(stream, hid, msg_W2, msg_b2, mmsg, CN_NM, CN_MSG, CN_HID, CN_HID, CN_MSG, false);
  launch_gemm(stream, mmsg, gru_Wih, gru_bih, gi, CN_NM, 1500, CN_MSG, CN_MSG, 1500, false);
  {
    const long nch = (long)CN_NM * (CN_D / 4);   // 16B chunks per row = 125
    k_gather_rows16<<<cdiv(nch, 256), 256, 0, stream>>>(h_g, memory, msg_nodes, nch, CN_D / 4);
  }
  launch_gemm(stream, h_g, gru_Whh, gru_bhh, gh, CN_NM, 1500, CN_D, CN_D, 1500, false);
  k_gru<<<cdiv((long)CN_NM * CN_D, 256), 256, 0, stream>>>(gi, gh, h_g, mem_upd);
  k_init_map<<<cdiv(CN_N, 256), 256, 0, stream>>>(map, CN_N);
  k_scatter_map<<<cdiv(CN_NM, 256), 256, 0, stream>>>(map, msg_nodes, CN_NM);

  // ============ Gather graph neighborhoods ============
  k_build_nodes<<<cdiv(CN_MQ1, 256), 256, 0, stream>>>(nodes, tsq, src_nodes, dst_nodes, neg_nodes, edge_times);
  k_gather_nbrs<<<cdiv(CN_MQ0, 256), 256, 0, stream>>>(nodes, CN_MQ0, nbr_nodes, nbr_eidx, nbr_times, n1f, e1f, t1f);
  k_gather_nbrs<<<cdiv(CN_R0, 256), 256, 0, stream>>>(n1f, CN_R0, nbr_nodes, nbr_eidx, nbr_times, n2f, e2f, t2f);

  // ============ Phase B0: layer-0 attention (queries = n1f) ============
  k_build_qin<<<cdiv((long)CN_MQ0 * CN_QD, 256), 256, 0, stream>>>(qin0, n1f, CN_MQ0, node_raw, memory, mem_upd, map, time_b);
  k_build_kin<<<cdiv((long)CN_R0 * CN_KD, 256), 256, 0, stream>>>(kin0, CN_R0, n2f, e2f, t2f, t1f, (const float*)nullptr,
      edge_raw, time_w, time_b, node_raw, memory, mem_upd, map);
  launch_gemm(stream, qin0, att_Wq, att_bq, q0, CN_MQ0, CN_QD, CN_QD, CN_QD, CN_QD, false);
  launch_gemm(stream, kin0, att_Wk, att_bk, kmat0, CN_R0, CN_QD, CN_KD, CN_KD, CN_QD, false);
  launch_gemm(stream, kin0, att_Wv, att_bv, vmat0, CN_R0, CN_QD, CN_KD, CN_KD, CN_QD, false);
  k_attn<<<cdiv((long)CN_MQ0 * 2, 8), 256, 0, stream>>>(q0, kmat0, vmat0, n2f, o0, CN_MQ0);
  launch_gemm(stream, o0, att_Wo, att_bo, hcat0, CN_MQ0, CN_QD, CN_QD, CN_QD, 1500, false);
  k_copy_cols<<<cdiv((long)CN_MQ0 * CN_D, 256), 256, 0, stream>>>(hcat0, 1500, CN_QD, qin0, CN_QD, CN_D, CN_MQ0);
  launch_gemm(stream, hcat0, mrg_W1, mrg_b1, mrgh0, CN_MQ0, CN_D, 1500, 1500, CN_D, true);
  launch_gemm(stream, mrgh0, mrg_W2, mrg_b2, emb1, CN_MQ0, CN_D, CN_D, CN_D, CN_D, false);

  // ============ Phase B1: layer-1 attention (queries = nodes) ============
  const float* Wq1 = att_Wq + (size_t)CN_QD * CN_QD;
  const float* Wk1 = att_Wk + (size_t)CN_KD * CN_QD;
  const float* Wv1 = att_Wv + (size_t)CN_KD * CN_QD;
  const float* Wo1 = att_Wo + (size_t)CN_QD * CN_QD;
  const float* mW1 = mrg_W1 + (size_t)1500 * CN_D;
  const float* mW2 = mrg_W2 + (size_t)CN_D * CN_D;
  k_build_qin<<<cdiv((long)CN_MQ1 * CN_QD, 256), 256, 0, stream>>>(qin1, nodes, CN_MQ1, node_raw, memory, mem_upd, map, time_b);
  k_build_kin<<<cdiv((long)CN_MQ0 * CN_KD, 256), 256, 0, stream>>>(kin1, CN_MQ0, n1f, e1f, t1f, tsq, emb1,
      edge_raw, time_w, time_b, node_raw, memory, mem_upd, map);
  launch_gemm(stream, qin1, Wq1, att_bq + CN_QD, q1, CN_MQ1, CN_QD, CN_QD, CN_QD, CN_QD, false);
  launch_gemm(stream, kin1, Wk1, att_bk + CN_QD, kmat1, CN_MQ0, CN_QD, CN_KD, CN_KD, CN_QD, false);
  launch_gemm(stream, kin1, Wv1, att_bv + CN_QD, vmat1, CN_MQ0, CN_QD, CN_KD, CN_KD, CN_QD, false);
  k_attn<<<cdiv((long)CN_MQ1 * 2, 8), 256, 0, stream>>>(q1, kmat1, vmat1, n1f, o1, CN_MQ1);
  launch_gemm(stream, o1, Wo1, att_bo + CN_QD, hcat1, CN_MQ1, CN_QD, CN_QD, CN_QD, 1500, false);
  k_copy_cols<<<cdiv((long)CN_MQ1 * CN_D, 256), 256, 0, stream>>>(hcat1, 1500, CN_QD, qin1, CN_QD, CN_D, CN_MQ1);
  launch_gemm(stream, hcat1, mW1, mrg_b1 + CN_D, mrgh1, CN_MQ1, CN_D, 1500, 1500, CN_D, true);
  launch_gemm(stream, mrgh1, mW2, mrg_b2 + CN_D, emb, CN_MQ1, CN_D, CN_D, CN_D, CN_D, false);

  // ============ Affinity head ============
  k_build_aff_in<<<cdiv((long)2 * CN_B * CN_QD, 256), 256, 0, stream>>>(Xaff, emb);
  launch_gemm(stream, Xaff, aff_W1, aff_b1, haff, 2 * CN_B, CN_D, CN_QD, CN_QD, CN_D, true);
  k_aff_out<<<cdiv((long)2 * CN_B, 8), 256, 0, stream>>>(haff, aff_W2, aff_b2, out, 2 * CN_B);
}
